// Ponder_AP_GCN_63857573757461
// MI455X (gfx1250) — matrix operand order repaired and checked by reading, not compiled, not run on hardware
//
#include <hip/hip_runtime.h>
#include <hip/hip_bf16.h>

// ---------------------------------------------------------------------------
// Ponder AP-GCN for MI455X (gfx1250, wave32).
//
// Strategy:
//  * Encoder GEMMs in bf16 WMMA (v_wmma_f32_16x16x32_bf16, f32 accum).
//    GEMM1: 100000x512x256 (26 GFLOP), GEMM2: 100000x256x64 (3.3 GFLOP).
//    Inputs/weights pre-packed into the ISA-documented 16x32 A / 32x16 B
//    per-lane fragment layouts so each WMMA operand is a single contiguous
//    32B load per lane (fully coalesced).
//  * GEMM1 epilogue (bias+relu) converts to bf16 and scatters straight into
//    GEMM2's A-fragment layout -> no repack pass over the 51MB hidden tensor.
//  * Propagation (dominant cost, ~1.7GB traffic/iter): gather ph[row]
//    (25.6MB state fits in 192MB L2) and hardware f32 atomic scatter to
//    ph2[col] (global_atomic_add_f32 via __hip_atomic_fetch_add relaxed/agent).
//    States ping-pong through the outputs[11,N,64] region of d_out.
//  * Halting head + sigmoid/cumprod epilogue are trivial element kernels.
// ---------------------------------------------------------------------------

typedef __attribute__((ext_vector_type(16))) __bf16 v16bf;
typedef __attribute__((ext_vector_type(8)))  float  v8f;

#define N_NODES   100000
#define N_EDGES   3200000
#define IN_F      512
#define HID       256
#define NCLS      64
#define NITER     10

#define MT   (N_NODES / 16)   // 6250 (exact)
#define KT1  (IN_F / 32)      // 16 k-steps GEMM1
#define NT1  (HID / 16)       // 16 n-tiles GEMM1
#define KT2  (HID / 32)       // 8 k-steps GEMM2
#define NT2  (NCLS / 16)      // 4 n-tiles GEMM2

#define FRAG_SHORTS 512       // 32 lanes * 16 bf16 per fragment
#define EPN (N_EDGES + N_NODES)

// d_out layout: outputs[11*N*64] ++ p[N*10] ++ halting_logits[N*10]
#define OUT_P_OFF  ((size_t)(NITER + 1) * N_NODES * NCLS)
#define OUT_H_OFF  (OUT_P_OFF + (size_t)N_NODES * NITER)

// workspace offsets (bytes)
#define WS_XPACK   ((size_t)0)                                   // 102,400,000
#define WS_W0PACK  (WS_XPACK  + (size_t)MT * KT1 * 1024)         //    262,144
#define WS_H1PACK  (WS_W0PACK + (size_t)NT1 * KT1 * 1024)        // 51,200,000
#define WS_W1PACK  (WS_H1PACK + (size_t)MT * KT2 * 1024)         //     32,768
#define WS_DINV    (WS_W1PACK + (size_t)NT2 * KT2 * 1024)        //    400,128
#define WS_NORM    (WS_DINV   + (size_t)400128)                  // 13,200,000

__device__ __forceinline__ unsigned short f2bf(float f) {
  unsigned int u = __float_as_uint(f);
  unsigned int r = u + 0x7FFFu + ((u >> 16) & 1u);   // round-to-nearest-even
  return (unsigned short)(r >> 16);
}

__device__ __forceinline__ void atomicAddF(float* p, float v) {
  __hip_atomic_fetch_add(p, v, __ATOMIC_RELAXED, __HIP_MEMORY_SCOPE_AGENT);
}

// ---- pack X (fp32 row-major) -> bf16 A-fragments (16x32) -------------------
// A layout: lane L: M = L%16, half h = L/16; pair j(0..3): K = 8h+2j;
//           pair j(4..7): K = 16 + 8h + 2(j-4).  elems 2j,2j+1 hold K,K+1.
__global__ __launch_bounds__(256) void pack_x_kernel(const float* __restrict__ x,
                                                     unsigned short* __restrict__ xp) {
  size_t tid = (size_t)blockIdx.x * blockDim.x + threadIdx.x;
  if (tid >= (size_t)MT * KT1 * 32 * 8) return;
  int frag = (int)(tid >> 8);
  int r    = (int)(tid & 255);
  int lane = r >> 3, j = r & 7;
  int mtile = frag / KT1, kt = frag % KT1;
  int m = mtile * 16 + (lane & 15);
  int h = lane >> 4;
  int kbase = (j < 4) ? (8 * h + 2 * j) : (16 + 8 * h + 2 * (j - 4));
  int k = kt * 32 + kbase;
  const float* src = x + (size_t)m * IN_F + k;
  size_t di = ((size_t)frag * 32 + lane) * 16 + 2 * j;
  xp[di]     = f2bf(src[0]);
  xp[di + 1] = f2bf(src[1]);
}

// ---- pack weight (fp32 KxN row-major) -> bf16 B-fragments (32x16) ----------
// B layout: lane L: N = L%16, half h = L/16; VGPR v(0..3) packs K = 8h+2v, +1;
//           v(4..7) packs K = 16 + 8h + 2(v-4), +1.
__global__ __launch_bounds__(256) void pack_w_kernel(const float* __restrict__ w,
                                                     unsigned short* __restrict__ wp,
                                                     int Kdim, int Ndim) {
  int nKt = Kdim / 32;
  size_t total = (size_t)nKt * (Ndim / 16) * 32 * 8;
  size_t tid = (size_t)blockIdx.x * blockDim.x + threadIdx.x;
  if (tid >= total) return;
  int frag = (int)(tid >> 8);
  int r    = (int)(tid & 255);
  int lane = r >> 3, v = r & 7;
  int ktile = frag % nKt, ntile = frag / nKt;
  int n = ntile * 16 + (lane & 15);
  int k = ktile * 32 + ((v < 4) ? (8 * (lane >> 4) + 2 * v) : (16 + 8 * (lane >> 4) + 2 * (v - 4)));
  size_t di = ((size_t)frag * 32 + lane) * 16 + 2 * v;
  wp[di]     = f2bf(w[(size_t)k * Ndim + n]);
  wp[di + 1] = f2bf(w[(size_t)(k + 1) * Ndim + n]);
}

// ---- GEMM1: H1 = relu(X @ W0 + b0), epilogue packs bf16 into GEMM2 A-frags -
__global__ __launch_bounds__(256) void gemm1_kernel(const unsigned short* __restrict__ xp,
                                                    const unsigned short* __restrict__ w0p,
                                                    const float* __restrict__ b0,
                                                    unsigned short* __restrict__ h1p) {
  int wave = threadIdx.x >> 5, lane = threadIdx.x & 31;
  int mtile = blockIdx.x * 8 + wave;
  if (mtile >= MT) return;
  int ntile = blockIdx.y;

  v8f acc = {};
  const v16bf* ap = reinterpret_cast<const v16bf*>(xp)  + ((size_t)mtile * KT1) * 32 + lane;
  const v16bf* bp = reinterpret_cast<const v16bf*>(w0p) + ((size_t)ntile * KT1) * 32 + lane;
#pragma unroll
  for (int kt = 0; kt < KT1; ++kt) {
    v16bf a = ap[(size_t)kt * 32];
    v16bf b = bp[(size_t)kt * 32];
    acc = __builtin_amdgcn_wmma_f32_16x16x32_bf16(false, a, false, b,
                                                  (short)0, acc, false, false);
  }

  int half = lane >> 4;
  int n = ntile * 16 + (lane & 15);       // GEMM1 col == GEMM2 A K-index
  float bias = b0[n];
  // inverse of the A-fragment layout for k_local = n%32
  int k_local = n & 31;
  int hh, j;
  if (k_local < 16) { hh = k_local >> 3; j = (k_local & 7) >> 1; }
  else { int kl = k_local - 16; hh = kl >> 3; j = 4 + ((kl & 7) >> 1); }
  int elem = 2 * j + (k_local & 1);
  int ktile2 = n >> 5;

#pragma unroll
  for (int r = 0; r < 8; ++r) {
    int m_loc = r + 8 * half;            // row within 16-row tile
    float vv = acc[r] + bias;
    vv = vv > 0.0f ? vv : 0.0f;
    size_t di = (((size_t)mtile * KT2 + ktile2) * 32 + (m_loc + 16 * hh)) * 16 + elem;
    h1p[di] = f2bf(vv);
  }
}

// ---- GEMM2: h = H1 @ W1 + b1 -> d_out[0 : N*64] (fp32) ---------------------
__global__ __launch_bounds__(256) void gemm2_kernel(const unsigned short* __restrict__ h1p,
                                                    const unsigned short* __restrict__ w1p,
                                                    const float* __restrict__ b1,
                                                    float* __restrict__ out) {
  int wave = threadIdx.x >> 5, lane = threadIdx.x & 31;
  int mtile = blockIdx.x * 8 + wave;
  if (mtile >= MT) return;
  int ntile = blockIdx.y;

  v8f acc = {};
  const v16bf* ap = reinterpret_cast<const v16bf*>(h1p) + ((size_t)mtile * KT2) * 32 + lane;
  const v16bf* bp = reinterpret_cast<const v16bf*>(w1p) + ((size_t)ntile * KT2) * 32 + lane;
#pragma unroll
  for (int kt = 0; kt < KT2; ++kt) {
    v16bf a = ap[(size_t)kt * 32];
    v16bf b = bp[(size_t)kt * 32];
    acc = __builtin_amdgcn_wmma_f32_16x16x32_bf16(false, a, false, b,
                                                  (short)0, acc, false, false);
  }

  int half = lane >> 4;
  int n = ntile * 16 + (lane & 15);
  float bias = b1[n];
#pragma unroll
  for (int r = 0; r < 8; ++r) {
    int m = mtile * 16 + r + 8 * half;
    out[(size_t)m * NCLS + n] = acc[r] + bias;
  }
}

// ---- degree / norm ---------------------------------------------------------
__global__ __launch_bounds__(256) void deg_init_kernel(float* __restrict__ deg) {
  int n = blockIdx.x * blockDim.x + threadIdx.x;
  if (n < N_NODES) deg[n] = 1.0f;                 // self-loop
}

__global__ __launch_bounds__(256) void deg_acc_kernel(const int* __restrict__ col,
                                                      float* __restrict__ deg) {
  int e = blockIdx.x * blockDim.x + threadIdx.x;
  if (e < N_EDGES) atomicAddF(&deg[col[e]], 1.0f);
}

__global__ __launch_bounds__(256) void dinv_kernel(float* __restrict__ deg) {
  int n = blockIdx.x * blockDim.x + threadIdx.x;
  if (n < N_NODES) {
    float d = deg[n];
    deg[n] = d > 0.0f ? rsqrtf(d) : 0.0f;
  }
}

__global__ __launch_bounds__(256) void norm_kernel(const int* __restrict__ row,
                                                   const int* __restrict__ col,
                                                   const float* __restrict__ dinv,
                                                   float* __restrict__ nrm) {
  int e = blockIdx.x * blockDim.x + threadIdx.x;
  if (e >= EPN) return;
  int r, c;
  if (e < N_EDGES) { r = row[e]; c = col[e]; }
  else             { r = c = e - N_EDGES; }
  nrm[e] = dinv[r] * dinv[c];
}

// ---- propagation: ph2[col] += norm * ph[row] (one thread per edge-feature) -
__global__ __launch_bounds__(256) void scatter_kernel(const float* __restrict__ ph,
                                                      float* __restrict__ ph2,
                                                      const int* __restrict__ row,
                                                      const int* __restrict__ col,
                                                      const float* __restrict__ nrm) {
  size_t tid = (size_t)blockIdx.x * blockDim.x + threadIdx.x;
  if (tid >= (size_t)EPN * NCLS) return;
  int e = (int)(tid >> 6);
  int c = (int)(tid & 63);
  int r, cc;
  if (e < N_EDGES) { r = row[e]; cc = col[e]; }
  else             { r = cc = e - N_EDGES; }
  float w = nrm[e];
  atomicAddF(&ph2[(size_t)cc * NCLS + c], w * ph[(size_t)r * NCLS + c]);
}

// ---- halting head: logit = clip(ph2 @ Wh + bh, -10, 10) --------------------
__global__ __launch_bounds__(256) void halting_kernel(const float* __restrict__ ph2,
                                                      const float* __restrict__ Wh,
                                                      const float* __restrict__ bh,
                                                      float* __restrict__ hlog, int t) {
  int n = blockIdx.x * blockDim.x + threadIdx.x;
  if (n >= N_NODES) return;
  const float4* p = reinterpret_cast<const float4*>(ph2 + (size_t)n * NCLS);
  const float4* w = reinterpret_cast<const float4*>(Wh);
  float s = bh[0];
#pragma unroll
  for (int i = 0; i < NCLS / 4; ++i) {
    float4 a = p[i], b = w[i];
    s += a.x * b.x + a.y * b.y + a.z * b.z + a.w * b.w;
  }
  s = fminf(fmaxf(s, -10.0f), 10.0f);
  hlog[(size_t)n * NITER + t] = s;
}

// ---- ponder probabilities --------------------------------------------------
__global__ __launch_bounds__(256) void p_kernel(const float* __restrict__ hlog,
                                                float* __restrict__ pout) {
  int n = blockIdx.x * blockDim.x + threadIdx.x;
  if (n >= N_NODES) return;
  float re = 1.0f;
  float pv[NITER];
#pragma unroll
  for (int i = 0; i < NITER; ++i) {
    float l = hlog[(size_t)n * NITER + i];
    float lam = 1.0f / (1.0f + expf(-l));
    pv[i] = lam * re;
    re *= (1.0f - lam);
  }
  pv[NITER - 1] += re;                    // leftover mass to last step
#pragma unroll
  for (int i = 0; i < NITER; ++i)
    pout[(size_t)n * NITER + i] = pv[i];
}

extern "C" void kernel_launch(void* const* d_in, const int* in_sizes, int n_in,
                              void* d_out, int out_size, void* d_ws, size_t ws_size,
                              hipStream_t stream) {
  const float* x   = (const float*)d_in[0];
  const int*   ei  = (const int*)d_in[1];     // [2, E] flat
  const float* W0  = (const float*)d_in[3];
  const float* b0  = (const float*)d_in[4];
  const float* W1  = (const float*)d_in[5];
  const float* b1  = (const float*)d_in[6];
  const float* Wh  = (const float*)d_in[7];
  const float* bh  = (const float*)d_in[8];
  const int* row = ei;
  const int* col = ei + N_EDGES;

  float* out = (float*)d_out;
  char*  ws  = (char*)d_ws;
  unsigned short* xpack  = (unsigned short*)(ws + WS_XPACK);
  unsigned short* w0pack = (unsigned short*)(ws + WS_W0PACK);
  unsigned short* h1pack = (unsigned short*)(ws + WS_H1PACK);
  unsigned short* w1pack = (unsigned short*)(ws + WS_W1PACK);
  float* dinv = (float*)(ws + WS_DINV);
  float* nrm  = (float*)(ws + WS_NORM);

  // --- encoder: pack + WMMA GEMMs ---
  {
    size_t t = (size_t)MT * KT1 * 32 * 8;
    pack_x_kernel<<<(unsigned)((t + 255) / 256), 256, 0, stream>>>(x, xpack);
  }
  pack_w_kernel<<<(KT1 * NT1 * 256 + 255) / 256, 256, 0, stream>>>(W0, w0pack, IN_F, HID);
  pack_w_kernel<<<(KT2 * NT2 * 256 + 255) / 256, 256, 0, stream>>>(W1, w1pack, HID, NCLS);

  dim3 g1((MT + 7) / 8, NT1);
  gemm1_kernel<<<g1, 256, 0, stream>>>(xpack, w0pack, b0, h1pack);
  dim3 g2((MT + 7) / 8, NT2);
  gemm2_kernel<<<g2, 256, 0, stream>>>(h1pack, w1pack, b1, out);  // outputs[0] = h

  // --- gcn_norm ---
  int nb = (N_NODES + 255) / 256;
  deg_init_kernel<<<nb, 256, 0, stream>>>(dinv);
  deg_acc_kernel<<<(N_EDGES + 255) / 256, 256, 0, stream>>>(col, dinv);
  dinv_kernel<<<nb, 256, 0, stream>>>(dinv);
  norm_kernel<<<(EPN + 255) / 256, 256, 0, stream>>>(row, col, dinv, nrm);

  // --- NITER propagation steps, states ping-pong through d_out outputs[] ---
  float* hlog = out + OUT_H_OFF;
  size_t segBytes = (size_t)N_NODES * NCLS * sizeof(float);
  unsigned scBlocks = (unsigned)(((size_t)EPN * NCLS + 255) / 256);
  for (int t = 0; t < NITER; ++t) {
    float* ph  = out + (size_t)t * N_NODES * NCLS;
    float* ph2 = out + (size_t)(t + 1) * N_NODES * NCLS;
    hipMemsetAsync(ph2, 0, segBytes, stream);
    scatter_kernel<<<scBlocks, 256, 0, stream>>>(ph, ph2, row, col, nrm);
    halting_kernel<<<nb, 256, 0, stream>>>(ph2, Wh, bh, hlog, t);
  }

  // --- ponder probabilities ---
  p_kernel<<<nb, 256, 0, stream>>>(hlog, out + OUT_P_OFF);
}
